// Head_50053548868112
// MI455X (gfx1250) — matrix operand order repaired and checked by reading, not compiled, not run on hardware
//
#include <hip/hip_runtime.h>

// ---------------------------------------------------------------------------
// Causal single-head attention for MI455X (gfx1250, wave32, WMMA).
//   Kernel 1: QKV projection, f16 WMMA 16x16x32, f32 accumulate.
//   Kernel 2: flash attention (online softmax), K/V staged in LDS,
//             S and P kept on-chip (never hit HBM).
// Async global->LDS staging of the K block (GLOBAL_LOAD_ASYNC_TO_LDS_B128 +
// s_wait_asynccnt) with corrected builtin signature:
//   (v4i AS1* gsrc, v4i AS3* lds, imm offset, imm cpol)
// ---------------------------------------------------------------------------

typedef __attribute__((ext_vector_type(16))) _Float16 v16h;
typedef __attribute__((ext_vector_type(8)))  _Float16 v8h;
typedef __attribute__((ext_vector_type(8)))  float    v8f;
typedef __attribute__((ext_vector_type(4)))  float    v4f;
typedef __attribute__((ext_vector_type(4)))  int      v4i;

#define WMMA_F16(A, B, C) \
  __builtin_amdgcn_wmma_f32_16x16x32_f16(false, (A), false, (B), (short)0, (C), false, false)

#if defined(__AMDGCN__) && __has_builtin(__builtin_amdgcn_global_load_async_to_lds_b128) && \
    __has_builtin(__builtin_amdgcn_s_wait_asynccnt)
#define HAVE_ASYNC_LDS 1
typedef __attribute__((address_space(1))) v4i* gptr_t;  // global <4 x i32>*
typedef __attribute__((address_space(3))) v4i* lptr_t;  // LDS    <4 x i32>*
#else
#define HAVE_ASYNC_LDS 0
#endif

static __device__ __forceinline__ v16h cat8(v8h lo, v8h hi) {
  return __builtin_shufflevector(lo, hi, 0, 1, 2, 3, 4, 5, 6, 7,
                                         8, 9, 10, 11, 12, 13, 14, 15);
}

#define N_EMBD 1024
#define HEAD   64
#define TSEQ   2048
#define NBATCH 8

// ---------------------------------------------------------------------------
// Kernel 1: Q/K/V = x @ Wq/Wk/Wv   (per-workgroup: 128 rows x 64 cols, x3)
// 256 threads = 8 waves; wave w owns M-tile w (16 rows), computes 4 N-tiles
// for each of the 3 weight matrices (12 accumulators of v8f).
// K loop: 32 chunks of K=32. x chunk and transposed W chunks staged in LDS
// as f16 (converted from the f32 inputs during staging).
// ---------------------------------------------------------------------------
__global__ __launch_bounds__(256) void qkv_proj_kernel(
    const float* __restrict__ x,
    const float* __restrict__ Wk, const float* __restrict__ Wq,
    const float* __restrict__ Wv,
    _Float16* __restrict__ Qh, _Float16* __restrict__ Kh,
    _Float16* __restrict__ Vh) {
  __shared__ __attribute__((aligned(32))) _Float16 xs[128 * 32];     // [row][k]
  __shared__ __attribute__((aligned(32))) _Float16 wt[3][64 * 32];   // [m][n][k] (transposed)

  const int tid  = threadIdx.x;
  const int lane = tid & 31;
  const int wave = tid >> 5;
  const int half = lane >> 4;
  const int l15  = lane & 15;
  const long rowBase = (long)blockIdx.x * 128;  // flat row in [0, B*T)

  const float* Wsrc[3] = {Wq, Wk, Wv};

  v8f acc[3][4];
#pragma unroll
  for (int m = 0; m < 3; ++m)
#pragma unroll
    for (int nt = 0; nt < 4; ++nt) acc[m][nt] = {};

  // staging roles (constant per thread)
  const int xrow = tid >> 1;         // 0..127
  const int xcol = (tid & 1) * 16;   // 0 or 16
  const int wn   = tid & 63;         // 0..63 : output column
  const int wk8  = (tid >> 6) * 8;   // 0,8,16,24 : k sub-block

  for (int kb = 0; kb < N_EMBD; kb += 32) {
    // ---- stage x[128][32] -> f16 LDS ----
    {
      const float* src = x + (rowBase + xrow) * N_EMBD + kb + xcol;
      v4f f0 = *(const v4f*)(src + 0);
      v4f f1 = *(const v4f*)(src + 4);
      v4f f2 = *(const v4f*)(src + 8);
      v4f f3 = *(const v4f*)(src + 12);
      if (kb + 32 < N_EMBD) __builtin_prefetch(src + 32, 0, 1);
      v8h h0, h1;
#pragma unroll
      for (int i = 0; i < 4; ++i) {
        h0[i]     = (_Float16)f0[i];
        h0[4 + i] = (_Float16)f1[i];
        h1[i]     = (_Float16)f2[i];
        h1[4 + i] = (_Float16)f3[i];
      }
      _Float16* dst = &xs[xrow * 32 + xcol];
      *(v8h*)(dst)     = h0;
      *(v8h*)(dst + 8) = h1;
    }
    // ---- stage W^T[64][32] x3 -> f16 LDS ----
#pragma unroll
    for (int m = 0; m < 3; ++m) {
      const float* wsrc = Wsrc[m] + (long)(kb + wk8) * HEAD + wn;
#pragma unroll
      for (int i = 0; i < 8; ++i)
        wt[m][wn * 32 + wk8 + i] = (_Float16)wsrc[i * HEAD];
    }
    __syncthreads();

    // ---- compute: A frag from xs, B frags from wt, 12 WMMAs ----
    const _Float16* xr = &xs[(wave * 16 + l15) * 32];
    v16h a = cat8(*(const v8h*)(xr + 16 * half),
                  *(const v8h*)(xr + 8 + 16 * half));
#pragma unroll
    for (int m = 0; m < 3; ++m) {
#pragma unroll
      for (int nt = 0; nt < 4; ++nt) {
        v16h b = *(const v16h*)(&wt[m][(nt * 16 + l15) * 32 + 16 * half]);
        acc[m][nt] = WMMA_F16(a, b, acc[m][nt]);
      }
    }
    __syncthreads();
  }

  // ---- epilogue: f32 accumulators -> f16 Q/K/V in workspace ----
  _Float16* Dst[3] = {Qh, Kh, Vh};
#pragma unroll
  for (int m = 0; m < 3; ++m) {
#pragma unroll
    for (int nt = 0; nt < 4; ++nt) {
#pragma unroll
      for (int r = 0; r < 8; ++r) {
        long row = rowBase + wave * 16 + r + 8 * half;
        Dst[m][row * HEAD + nt * 16 + l15] = (_Float16)acc[m][nt][r];
      }
    }
  }
}

// ---------------------------------------------------------------------------
// Kernel 2: flash attention. Grid = B x (T/128). 8 waves; wave w owns the
// 16-query tile starting at q0 + 16*w. Loop over 32-key blocks up to the
// workgroup's causal limit; K block (async DMA to LDS when available) and
// V^T block staged in LDS and shared by all 8 waves. Online softmax per row;
// P transposed through a per-wave LDS scratch to build the WMMA A fragment
// for P·V.
// ---------------------------------------------------------------------------
__global__ __launch_bounds__(256) void attn_kernel(
    const _Float16* __restrict__ Qh, const _Float16* __restrict__ Kh,
    const _Float16* __restrict__ Vh, float* __restrict__ out) {
  __shared__ __attribute__((aligned(32))) _Float16 Ks[32 * 64];      // [key][dim]
  __shared__ __attribute__((aligned(32))) _Float16 Vt[64 * 32];      // [dim][key]
  __shared__ __attribute__((aligned(32))) _Float16 Ps[8][16 * 32];   // per-wave P tile

  const int tid  = threadIdx.x;
  const int lane = tid & 31;
  const int wave = tid >> 5;
  const int half = lane >> 4;
  const int l15  = lane & 15;

  const int  bId  = blockIdx.x >> 4;
  const int  qblk = blockIdx.x & 15;
  const int  q0   = qblk * 128;
  const long base = (long)bId * TSEQ * HEAD;
  const _Float16* Qb = Qh + base;
  const _Float16* Kb = Kh + base;
  const _Float16* Vb = Vh + base;
  float* Ob = out + base;

  // Q tile A-fragments (held for the whole kernel): 16 rows x 64 dims
  const int qrow = q0 + wave * 16 + l15;
  const _Float16* qptr = Qb + (long)qrow * HEAD;
  const v16h qa0 = cat8(*(const v8h*)(qptr + 16 * half),
                        *(const v8h*)(qptr + 8 + 16 * half));
  const v16h qa1 = cat8(*(const v8h*)(qptr + 32 + 16 * half),
                        *(const v8h*)(qptr + 40 + 16 * half));

  v8f o[4];
#pragma unroll
  for (int nt = 0; nt < 4; ++nt) o[nt] = {};
  float mrow[8], lrow[8];
#pragma unroll
  for (int r = 0; r < 8; ++r) { mrow[r] = -1e30f; lrow[r] = 0.0f; }

  // staging roles
  const int skey = tid >> 3;        // 0..31
  const int sd   = (tid & 7) * 8;   // 0..56

  const int nblk = (q0 + 128) >> 5;  // causal limit for this workgroup
  for (int blk = 0; blk < nblk; ++blk) {
    const int kb0 = blk * 32;

    // ---- stage K[32][64] (async DMA when available) and V^T[64][32] ----
    {
      const _Float16* ksrc = Kb + (long)(kb0 + skey) * HEAD + sd;
#if HAVE_ASYNC_LDS
      __builtin_amdgcn_global_load_async_to_lds_b128(
          (gptr_t)(void*)ksrc, (lptr_t)(void*)&Ks[skey * 64 + sd], 0, 0);
#else
      v8h kf = *(const v8h*)ksrc;
      *(v8h*)(&Ks[skey * 64 + sd]) = kf;
#endif
      v8h vf = *(const v8h*)(Vb + (long)(kb0 + skey) * HEAD + sd);
#pragma unroll
      for (int i = 0; i < 8; ++i) Vt[(sd + i) * 32 + skey] = vf[i];

      // prefetch next key block while we compute on this one
      if (blk + 1 < nblk) {
        __builtin_prefetch(Kb + (long)(kb0 + 32 + skey) * HEAD + sd, 0, 1);
        __builtin_prefetch(Vb + (long)(kb0 + 32 + skey) * HEAD + sd, 0, 1);
      }
    }
#if HAVE_ASYNC_LDS
    __builtin_amdgcn_s_wait_asynccnt(0);
#endif
    __syncthreads();

    // ---- S = Q K^T for the two 16-key subtiles (K-dim split 0..31 / 32..63)
    v8f st[2];
#pragma unroll
    for (int kt = 0; kt < 2; ++kt) {
      v8f s = {};
      const _Float16* kr = &Ks[(kt * 16 + l15) * 64];
      v16h b0 = *(const v16h*)(kr + 16 * half);
      v16h b1 = *(const v16h*)(kr + 32 + 16 * half);
      s = WMMA_F16(qa0, b0, s);
      s = WMMA_F16(qa1, b1, s);
      st[kt] = s;
    }

    // ---- online softmax + write P (f16) into per-wave LDS scratch ----
    _Float16* pw = &Ps[wave][0];
#pragma unroll
    for (int r = 0; r < 8; ++r) {
      const int qi = q0 + wave * 16 + r + 8 * half;
      float s0v = st[0][r] * 0.125f;  // 1/sqrt(64)
      float s1v = st[1][r] * 0.125f;
      s0v = (kb0 + l15 > qi) ? -1e30f : s0v;
      s1v = (kb0 + 16 + l15 > qi) ? -1e30f : s1v;
      float mx = fmaxf(s0v, s1v);
      mx = fmaxf(mx, __shfl_xor(mx, 1, 32));
      mx = fmaxf(mx, __shfl_xor(mx, 2, 32));
      mx = fmaxf(mx, __shfl_xor(mx, 4, 32));
      mx = fmaxf(mx, __shfl_xor(mx, 8, 32));
      const float mnew  = fmaxf(mrow[r], mx);
      const float alpha = __expf(mrow[r] - mnew);
      mrow[r] = mnew;
      const float p0 = __expf(s0v - mnew);
      const float p1 = __expf(s1v - mnew);
      float ps = p0 + p1;
      ps += __shfl_xor(ps, 1, 32);
      ps += __shfl_xor(ps, 2, 32);
      ps += __shfl_xor(ps, 4, 32);
      ps += __shfl_xor(ps, 8, 32);
      lrow[r] = lrow[r] * alpha + ps;
#pragma unroll
      for (int nt = 0; nt < 4; ++nt) o[nt][r] *= alpha;
      pw[(r + 8 * half) * 32 + l15]      = (_Float16)p0;
      pw[(r + 8 * half) * 32 + 16 + l15] = (_Float16)p1;
    }
    // same-wave LDS store->load is in-order in HW; fence stops compiler reorder
    asm volatile("" ::: "memory");

    // ---- P (A-frag layout) from LDS, then O += P · V (4 WMMAs, K=32) ----
    v16h pa = cat8(*(const v8h*)(&pw[l15 * 32 + 16 * half]),
                   *(const v8h*)(&pw[l15 * 32 + 8 + 16 * half]));
#pragma unroll
    for (int nt = 0; nt < 4; ++nt) {
      v16h vb = *(const v16h*)(&Vt[(nt * 16 + l15) * 32 + 16 * half]);
      o[nt] = WMMA_F16(pa, vb, o[nt]);
    }
    __syncthreads();
  }

  // ---- epilogue: normalize by row sum, store f32 output ----
#pragma unroll
  for (int nt = 0; nt < 4; ++nt) {
#pragma unroll
    for (int r = 0; r < 8; ++r) {
      const int row = q0 + wave * 16 + r + 8 * half;
      Ob[(long)row * HEAD + nt * 16 + l15] = o[nt][r] / lrow[r];
    }
  }
}

// ---------------------------------------------------------------------------
// Host-side launcher
// ---------------------------------------------------------------------------
extern "C" void kernel_launch(void* const* d_in, const int* in_sizes, int n_in,
                              void* d_out, int out_size, void* d_ws,
                              size_t ws_size, hipStream_t stream) {
  const float* x  = (const float*)d_in[0];
  const float* Wk = (const float*)d_in[1];
  const float* Wq = (const float*)d_in[2];
  const float* Wv = (const float*)d_in[3];
  float* out = (float*)d_out;

  // workspace: Q,K,V in f16, each B*T*HEAD elements (6 MB total)
  _Float16* Qh = (_Float16*)d_ws;
  _Float16* Kh = Qh + (size_t)NBATCH * TSEQ * HEAD;
  _Float16* Vh = Kh + (size_t)NBATCH * TSEQ * HEAD;

  dim3 blk(256);
  dim3 grid1((NBATCH * TSEQ) / 128);  // 128 workgroups
  hipLaunchKernelGGL(qkv_proj_kernel, grid1, blk, 0, stream,
                     x, Wk, Wq, Wv, Qh, Kh, Vh);

  dim3 grid2(NBATCH * (TSEQ / 128));  // 128 workgroups
  hipLaunchKernelGGL(attn_kernel, grid2, blk, 0, stream, Qh, Kh, Vh, out);
}
